// EAKT_base_16441134809606
// MI455X (gfx1250) — matrix operand order repaired and checked by reading, not compiled, not run on hardware
//
#include <hip/hip_runtime.h>
#include <hip/hip_bf16.h>
#include <stdint.h>

#define E 256
#define NQ 50
#define NHID 64
#define NHEADS 4
#define B 256
#define L 200
#define G 1024           // 4*E gate width
#define CAT 1536         // 6*E
#define TABROWS 135      // 51+50+20+6+6+2

#define KLDS 128         // K-half of W_hh staged in LDS
// 256 KB W_hh-lo + 2 x 8 KB h double-buffer = 272 KB (< 320 KB WGP LDS)
#define LSTM_SMEM (G * KLDS * 2 + 2 * 16 * E * 2)

#if __has_builtin(__builtin_amdgcn_sched_group_barrier)
#define SCHED_GROUP(mask, size, id) __builtin_amdgcn_sched_group_barrier(mask, size, id)
#else
#define SCHED_GROUP(mask, size, id)
#endif

typedef float  v8f  __attribute__((ext_vector_type(8)));
typedef __bf16 v16bf __attribute__((ext_vector_type(16)));

union Ubf { uint4 q[2]; v16bf v; };

__device__ __forceinline__ float sigmf(float x) { return 1.0f / (1.0f + __expf(-x)); }

// ---------------------------------------------------------------------------
// K1: GAT (50 nodes, 4 heads) -> q_emb_all (50 x 256). Single block, scalar.
// ---------------------------------------------------------------------------
__global__ void k_gat(const int* __restrict__ d_all_idx, const int* __restrict__ adj,
                      const float* __restrict__ emb_diff,
                      const float* __restrict__ gat_W, const float* __restrict__ gat_a,
                      const float* __restrict__ gat_out_W, const float* __restrict__ gat_out_b,
                      float* __restrict__ ws_qemb, float* __restrict__ ws_dall,
                      float* __restrict__ ws_Wh, float* __restrict__ ws_e,
                      float* __restrict__ ws_x)
{
    const int tid = threadIdx.x;  // 256 threads
    for (int i = tid; i < NQ * E; i += 256) {
        int row = i >> 8, col = i & 255;
        ws_dall[i] = emb_diff[d_all_idx[row] * E + col];
    }
    __syncthreads();
    __shared__ float sWh1[NQ], sWh2[NQ];
    for (int hd = 0; hd < NHEADS; ++hd) {
        const float* W = gat_W + (size_t)hd * E * NHID;
        const float* a = gat_a + hd * 2 * NHID;
        for (int i = tid; i < NQ * NHID; i += 256) {
            int row = i / NHID, col = i % NHID;
            float acc = 0.f;
            for (int k = 0; k < E; ++k) acc += ws_dall[row * E + k] * W[k * NHID + col];
            ws_Wh[i] = acc;
        }
        __syncthreads();
        if (tid < NQ) {
            float a1 = 0.f, a2 = 0.f;
            for (int c = 0; c < NHID; ++c) {
                a1 += ws_Wh[tid * NHID + c] * a[c];
                a2 += ws_Wh[tid * NHID + c] * a[NHID + c];
            }
            sWh1[tid] = a1; sWh2[tid] = a2;
        }
        __syncthreads();
        if (tid < NQ) {
            float tmp[NQ];
            float mx = -3.0e38f;
            for (int j = 0; j < NQ; ++j) {
                float ev = sWh1[tid] + sWh2[j];
                ev = ev > 0.f ? ev : 0.2f * ev;
                ev = (adj[tid * NQ + j] > 0) ? ev : -9.0e15f;
                tmp[j] = ev; mx = fmaxf(mx, ev);
            }
            float ssum = 0.f;
            for (int j = 0; j < NQ; ++j) { float e2 = __expf(tmp[j] - mx); tmp[j] = e2; ssum += e2; }
            float inv = 1.0f / ssum;
            for (int j = 0; j < NQ; ++j) ws_e[tid * NQ + j] = tmp[j] * inv;
        }
        __syncthreads();
        for (int i = tid; i < NQ * NHID; i += 256) {
            int row = i / NHID, col = i % NHID;
            float acc = 0.f;
            for (int j = 0; j < NQ; ++j) acc += ws_e[row * NQ + j] * ws_Wh[j * NHID + col];
            acc = acc > 0.f ? acc : (__expf(acc) - 1.0f);
            ws_x[row * E + hd * NHID + col] = acc;
        }
        __syncthreads();
    }
    for (int i = tid; i < NQ * E; i += 256) {
        int row = i >> 8, col = i & 255;
        float acc = gat_out_b[col];
        for (int k = 0; k < E; ++k) acc += ws_x[row * E + k] * gat_out_W[col * E + k];
        ws_qemb[i] = acc > 0.f ? acc : 0.2f * acc;
    }
}

// ---------------------------------------------------------------------------
// K2: project every embedding-table row through its W_ih segment.
// ---------------------------------------------------------------------------
__global__ void k_tabproj(const float* __restrict__ emb_q, const float* __restrict__ qemb_all,
                          const float* __restrict__ emb_group, const float* __restrict__ emb_time,
                          const float* __restrict__ emb_skill, const float* __restrict__ emb_ans,
                          const float* __restrict__ W_ih, float* __restrict__ tab)
{
    int idx = blockIdx.x * blockDim.x + threadIdx.x;
    if (idx >= TABROWS * G) return;
    int row = idx >> 10, n = idx & 1023;
    const float* src; int seg;
    if      (row < 51)  { src = emb_q     + (size_t)row          * E; seg = 0; }
    else if (row < 101) { src = qemb_all  + (size_t)(row - 51)   * E; seg = 1; }
    else if (row < 121) { src = emb_group + (size_t)(row - 101)  * E; seg = 2; }
    else if (row < 127) { src = emb_time  + (size_t)(row - 121)  * E; seg = 3; }
    else if (row < 133) { src = emb_skill + (size_t)(row - 127)  * E; seg = 4; }
    else                { src = emb_ans   + (size_t)(row - 133)  * E; seg = 5; }
    const float* w = W_ih + (size_t)n * CAT + seg * E;
    float acc = 0.f;
    for (int k = 0; k < E; ++k) acc += src[k] * w[k];
    tab[idx] = acc;
}

// ---------------------------------------------------------------------------
// K3: W_hh (1024 x 256) fp32 -> bf16
// ---------------------------------------------------------------------------
__global__ void k_whh_bf16(const float* __restrict__ whh, __bf16* __restrict__ out)
{
    int i = blockIdx.x * 256 + threadIdx.x;
    if (i < G * E) out[i] = (__bf16)whh[i];
}

// ---------------------------------------------------------------------------
// K4: gather-sum: x_proj[t][b][n] = sum_seg tab[row_seg][n] + b_ih[n] + b_hh[n]
// ---------------------------------------------------------------------------
__global__ void k_gather(const int* __restrict__ q, const int* __restrict__ r,
                         const int* __restrict__ s, const int* __restrict__ g,
                         const int* __restrict__ t, const float* __restrict__ tab,
                         const float* __restrict__ b_ih, const float* __restrict__ b_hh,
                         float* __restrict__ xproj)
{
    int bl = blockIdx.x;
    int b = bl / L, l = bl % L;
    int ii = b * L + l;
    __shared__ int rows[6];
    if (threadIdx.x == 0) {
        int qi = q[ii];
        int qm = qi - 1; if (qm < 0) qm = 0;
        rows[0] = qi;
        rows[1] = 51 + qm;
        rows[2] = 101 + g[ii];
        rows[3] = 121 + t[ii];
        rows[4] = 127 + s[ii];
        rows[5] = 133 + r[ii];
    }
    __syncthreads();
    float* dst = xproj + ((size_t)l * B + b) * G;
    for (int n = threadIdx.x; n < G; n += 256) {
        float acc = b_ih[n] + b_hh[n];
#pragma unroll
        for (int sgi = 0; sgi < 6; ++sgi) acc += tab[(size_t)rows[sgi] * G + n];
        dst[n] = acc;
    }
}

// ---------------------------------------------------------------------------
// K5: LSTM recurrence with WMMA. 16 blocks x 256 threads (8 wave32).
// - Lower K-half of W_hh staged in LDS via global_load_async_to_lds_b128.
// - h double-buffered in LDS -> single barrier per step.
// - Per ks: sched_group_barrier pins "8 DS/VMEM reads then 4 WMMAs" so the
//   scheduler cannot re-serialize into one B buffer with a full wait per
//   WMMA (rounds 3/4 issue) -> partial counter waits, pipelined loads.
// ---------------------------------------------------------------------------
__global__ void __launch_bounds__(256) k_lstm(const float* __restrict__ xproj,
                                              const __bf16* __restrict__ whh,
                                              float* __restrict__ h_all)
{
    extern __shared__ char smem[];
    __bf16* s_w  = (__bf16*)smem;                        // 1024 x 128 bf16, 256 KB
    __bf16* s_h0 = (__bf16*)(smem + G * KLDS * 2);       // 16 x 256 bf16, 8 KB
    __bf16* s_h1 = s_h0 + 16 * E;                        // 16 x 256 bf16, 8 KB

    const int tid  = threadIdx.x;
    const int lane = tid & 31, wave = tid >> 5;
    const int m16  = lane & 15, half = lane >> 4;
    const int wgbase = blockIdx.x * 16;
    const int nbase  = wave * 32;

    // ---- async stage of W_hh[:, 0:128) into LDS (16 B chunks, ASYNCcnt) ----
    {
        const uint32_t s_w_base = (uint32_t)(uintptr_t)s_w;
        for (int ci = tid; ci < G * 16; ci += 256) {        // 16384 x 16B
            int n = ci >> 4, u = ci & 15;
            uint32_t goff  = (uint32_t)(n * (E * 2) + u * 16);   // lower-half bytes
            uint32_t laddr = s_w_base + (uint32_t)(n * (KLDS * 2) + u * 16);
            asm volatile("global_load_async_to_lds_b128 %0, %1, %2"
                         :: "v"(laddr), "v"(goff), "s"(whh)
                         : "memory");
        }
    }
    for (int i = tid; i < 16 * E; i += 256) s_h0[i] = (__bf16)0.0f;
    v8f cst[2];
#pragma unroll
    for (int jt = 0; jt < 2; ++jt)
#pragma unroll
        for (int rr = 0; rr < 8; ++rr) cst[jt][rr] = 0.0f;
    asm volatile("s_wait_asynccnt 0" ::: "memory");
    __syncthreads();

    for (int t = 0; t < L; ++t) {
        const __bf16* hcur = (t & 1) ? s_h1 : s_h0;
        __bf16*       hnxt = (t & 1) ? s_h0 : s_h1;

        // prefetch next step's x_proj slab (64 KB) into cache
        const float* xpn = xproj + ((size_t)((t + 1 < L) ? t + 1 : t) * B + wgbase) * G;
        __builtin_prefetch(xpn + tid * 32, 0, 3);
        __builtin_prefetch(xpn + 8192 + tid * 32, 0, 3);

        // ---- A fragments (16x32 bf16, contiguous k = 16*half + s) from current-h buffer ----
        v16bf afrag[8];
#pragma unroll
        for (int ks = 0; ks < 8; ++ks) {
#pragma unroll
            for (int v = 0; v < 8; ++v) {
                int kb = ks * 32 + half * 16 + ((v < 4) ? 0 : 8) + (v & 3) * 2;
                afrag[ks][2 * v]     = hcur[m16 * E + kb];
                afrag[ks][2 * v + 1] = hcur[m16 * E + kb + 1];
            }
        }

        const float* xp = xproj + ((size_t)t * B + wgbase) * G;

        // ---- process each hidden-half: 4 gate tiles live at a time ----
#pragma unroll
        for (int jt = 0; jt < 2; ++jt) {
            v8f acc[4];
            // init accumulators with x_proj (C/D layout: lane=col, VGPR=row)
#pragma unroll
            for (int gate = 0; gate < 4; ++gate) {
                int n0 = gate * E + nbase + jt * 16 + m16;
#pragma unroll
                for (int rr = 0; rr < 8; ++rr)
                    acc[gate][rr] = xp[(size_t)(rr + half * 8) * G + n0];
            }
            // K loop: ks outer, gates inner. Per ks: 8 b128 loads into 4
            // distinct buffers, then 4 independent WMMAs (distinct acc).
#pragma unroll
            for (int ks = 0; ks < 8; ++ks) {
                asm volatile("" ::: "memory");   // bound in-flight B; stop t-loop hoisting
                Ubf bufs[4];
#pragma unroll
                for (int gate = 0; gate < 4; ++gate) {
                    int ncol = gate * E + nbase + jt * 16 + m16;
                    if (ks < 4) {
                        const uint4* wl = (const uint4*)(s_w + (size_t)ncol * KLDS + half * 16);
                        bufs[gate].q[0] = wl[ks * 4];
                        bufs[gate].q[1] = wl[ks * 4 + 1];
                    } else {
                        const uint4* wg = (const uint4*)(whh + (size_t)ncol * E + half * 16);
                        bufs[gate].q[0] = wg[ks * 4];
                        bufs[gate].q[1] = wg[ks * 4 + 1];
                    }
                }
#pragma unroll
                for (int gate = 0; gate < 4; ++gate)
                    acc[gate] = __builtin_amdgcn_wmma_f32_16x16x32_bf16(
                        false, afrag[ks], false, bufs[gate].v, (short)0, acc[gate], false, false);
                // pin schedule for this region: 8 reads first, then 4 WMMAs
                SCHED_GROUP(0x120, 8, 0);   // DS-read | VMEM-read group
                SCHED_GROUP(0x008, 4, 0);   // MFMA/WMMA group
            }
            // wave-local gate nonlinearity + cell update for this hidden-half
            int j = nbase + jt * 16 + m16;
#pragma unroll
            for (int rr = 0; rr < 8; ++rr) {
                float iv = sigmf(acc[0][rr]);
                float fv = sigmf(acc[1][rr]);
                float gv = tanhf(acc[2][rr]);
                float ov = sigmf(acc[3][rr]);
                float c = fv * cst[jt][rr] + iv * gv;
                cst[jt][rr] = c;
                float h = ov * tanhf(c);
                int mrow = rr + half * 8;
                h_all[(((size_t)(wgbase + mrow)) * L + t) * E + j] = h;
                hnxt[mrow * E + j] = (__bf16)h;
            }
        }
        __syncthreads();   // next-h buffer complete before next step reads it
    }
}

// ---------------------------------------------------------------------------
// K6: y = sigmoid(h @ out_W.T + out_b), fused trim/replicate.
// ---------------------------------------------------------------------------
__global__ void k_out(const float* __restrict__ h_all, const float* __restrict__ out_W,
                      const float* __restrict__ out_b, float* __restrict__ y)
{
    size_t idx = (size_t)blockIdx.x * blockDim.x + threadIdx.x;
    if (idx >= (size_t)B * L * NQ) return;
    int qo = (int)(idx % NQ);
    size_t bl = idx / NQ;
    int l = (int)(bl % L);
    int b = (int)(bl / L);
    int sl = (l < 190) ? l : 39;
    const float* h = h_all + ((size_t)b * L + sl) * E;
    const float* w = out_W + (size_t)qo * E;
    float acc = out_b[qo];
    for (int k = 0; k < E; ++k) acc += h[k] * w[k];
    y[idx] = sigmf(acc);
}

// ---------------------------------------------------------------------------
extern "C" void kernel_launch(void* const* d_in, const int* in_sizes, int n_in,
                              void* d_out, int out_size, void* d_ws, size_t ws_size,
                              hipStream_t stream)
{
    const int*   q         = (const int*)  d_in[0];
    const int*   r         = (const int*)  d_in[1];
    // d_in[2] ('d') is unused by the reference
    const int*   s         = (const int*)  d_in[3];
    const int*   g         = (const int*)  d_in[4];
    const int*   t         = (const int*)  d_in[5];
    const int*   d_all_idx = (const int*)  d_in[6];
    const int*   adj       = (const int*)  d_in[7];
    const float* emb_q     = (const float*)d_in[8];
    const float* emb_ans   = (const float*)d_in[9];
    const float* emb_diff  = (const float*)d_in[10];
    const float* emb_skill = (const float*)d_in[11];
    const float* emb_group = (const float*)d_in[12];
    const float* emb_time  = (const float*)d_in[13];
    const float* gat_W     = (const float*)d_in[14];
    const float* gat_a     = (const float*)d_in[15];
    const float* gat_out_W = (const float*)d_in[16];
    const float* gat_out_b = (const float*)d_in[17];
    const float* W_ih      = (const float*)d_in[18];
    const float* W_hh      = (const float*)d_in[19];
    const float* b_ih      = (const float*)d_in[20];
    const float* b_hh      = (const float*)d_in[21];
    const float* out_W     = (const float*)d_in[22];
    const float* out_b     = (const float*)d_in[23];
    float* y = (float*)d_out;

    char* base = (char*)d_ws;
    size_t off = 0;
    auto carve = [&](size_t bytes) -> char* {
        char* p = base + off;
        off = (off + bytes + 255) & ~(size_t)255;
        return p;
    };
    float*  ws_qemb = (float*) carve((size_t)NQ * E * 4);
    float*  ws_dall = (float*) carve((size_t)NQ * E * 4);
    float*  ws_Wh   = (float*) carve((size_t)NQ * NHID * 4);
    float*  ws_e    = (float*) carve((size_t)NQ * NQ * 4);
    float*  ws_x    = (float*) carve((size_t)NQ * E * 4);
    float*  ws_tab  = (float*) carve((size_t)TABROWS * G * 4);
    __bf16* ws_whh  = (__bf16*)carve((size_t)G * E * 2);
    float*  ws_xp   = (float*) carve((size_t)L * B * G * 4);     // ~210 MB
    float*  ws_h    = (float*) carve((size_t)B * L * E * 4);     // ~52 MB
    (void)ws_size; (void)in_sizes; (void)n_in; (void)out_size;

    k_gat<<<1, 256, 0, stream>>>(d_all_idx, adj, emb_diff, gat_W, gat_a,
                                 gat_out_W, gat_out_b,
                                 ws_qemb, ws_dall, ws_Wh, ws_e, ws_x);
    k_tabproj<<<(TABROWS * G + 255) / 256, 256, 0, stream>>>(
        emb_q, ws_qemb, emb_group, emb_time, emb_skill, emb_ans, W_ih, ws_tab);
    k_whh_bf16<<<(G * E + 255) / 256, 256, 0, stream>>>(W_hh, ws_whh);
    k_gather<<<B * L, 256, 0, stream>>>(q, r, s, g, t, ws_tab, b_ih, b_hh, ws_xp);
    k_lstm<<<B / 16, 256, LSTM_SMEM, stream>>>(ws_xp, ws_whh, ws_h);
    k_out<<<((size_t)B * L * NQ + 255) / 256, 256, 0, stream>>>(ws_h, out_W, out_b, y);
}
